// PointRendMaskHead_20298015441303
// MI455X (gfx1250) — matrix operand order repaired and checked by reading, not compiled, not run on hardware
//
#include <hip/hip_runtime.h>

typedef _Float16 h16;
typedef __attribute__((ext_vector_type(8)))  _Float16 v8h;
typedef __attribute__((ext_vector_type(16))) _Float16 v16h;
typedef __attribute__((ext_vector_type(8)))  float    v8f;

// Problem constants
#define R_   512
#define C_   256
#define HF_  192
#define WF_  288
#define HW_  (HF_*WF_)     // 55296
#define POOL_ 14
#define OUTRES_ 7
#define NCLS_ 80
#define NSAMP_ 588
#define NUNC_ 147
#define NRAND_ 49
#define NPTS_ 196
#define FCDIM_ 1024
#define FCIN_  (C_*OUTRES_*OUTRES_)    // 12544
#define PREDN_ (NCLS_*OUTRES_*OUTRES_) // 3920
#define PREDNP_ 4096                   // padded to multiple of 256
#define PHIN_  336
#define PHINP_ 352                     // padded to multiple of 32
#define PHM_   (R_*NPTS_)              // 100352

enum { MODE_ROW_F16 = 0, MODE_CONV = 1, MODE_ROW_F32 = 2, MODE_OUT = 3 };

__device__ __forceinline__ v8f wmma_f16(v16h a, v16h b, v8f c) {
  // D = A(16x32 f16) * B(32x16 f16) + C(16x16 f32)
  return __builtin_amdgcn_wmma_f32_16x16x32_f16(false, a, false, b, (short)0, c,
                                                false, false);
}

// A-fragment per-lane load: 8 halves at p, 8 halves at p+16 (ISA 16x32 layout)
__device__ __forceinline__ v16h load_afrag(const h16* __restrict__ p) {
  v8h l = *(const v8h*)(p);
  v8h h = *(const v8h*)(p + 16);
  v16h a;
#pragma unroll
  for (int i = 0; i < 8; ++i) { a[i] = l[i]; a[8 + i] = h[i]; }
  return a;
}

// Zero-padded bilinear sample matching F.grid_sample(align_corners=False, zeros)
__device__ __forceinline__ float bilin(const float* __restrict__ pl, int H, int W,
                                       float x, float y) {
  float x0f = floorf(x), y0f = floorf(y);
  int ix = (int)x0f, iy = (int)y0f;
  float wx1 = x - x0f, wy1 = y - y0f;
  float wx0 = 1.0f - wx1, wy0 = 1.0f - wy1;
  float v = 0.0f;
  bool vx0 = (ix >= 0) && (ix < W);
  bool vx1 = (ix + 1 >= 0) && (ix + 1 < W);
  if (iy >= 0 && iy < H) {
    if (vx0) v += wx0 * wy0 * pl[iy * W + ix];
    if (vx1) v += wx1 * wy0 * pl[iy * W + ix + 1];
  }
  if (iy + 1 >= 0 && iy + 1 < H) {
    if (vx0) v += wx0 * wy1 * pl[(iy + 1) * W + ix];
    if (vx1) v += wx1 * wy1 * pl[(iy + 1) * W + ix + 1];
  }
  return v;
}

// ------------- weight conversion f32 -> f16 with N-row and K-col zero pad --
__global__ void cvt_pad_kernel(const float* __restrict__ src, h16* __restrict__ dst,
                               int N, int K, int Kp, long total) {
  long i = (long)blockIdx.x * blockDim.x + threadIdx.x;
  if (i >= total) return;
  int kp = (int)(i % Kp);
  long n = i / Kp;
  dst[i] = (n < N && kp < K) ? (h16)src[n * K + kp] : (h16)0.0f;
}

// ---------------- K1: ROI 14x14 bilinear sampling -> im2col for conv -------
// Aconv[(r*49 + oy*7 + ox)][ic*4 + ky*2 + kx], K = 1024
__global__ __launch_bounds__(256) void roi_im2col_kernel(
    const float* __restrict__ feat, const float* __restrict__ boxes,
    h16* __restrict__ Aconv) {
  int bid = blockIdx.x;          // r*196 + p
  int r = bid / (POOL_ * POOL_);
  int p = bid % (POOL_ * POOL_);
  int py = p / POOL_, px = p % POOL_;
  float bx0 = boxes[r * 4 + 0], by0 = boxes[r * 4 + 1];
  float bx1 = boxes[r * 4 + 2], by1 = boxes[r * 4 + 3];
  float gx = (px + 0.5f) * (1.0f / POOL_);
  float gy = (py + 0.5f) * (1.0f / POOL_);
  float fx = (bx0 + gx * (bx1 - bx0)) * 0.25f - 0.5f;  // feature-space coord
  float fy = (by0 + gy * (by1 - by0)) * 0.25f - 0.5f;
  int ch = threadIdx.x;
  float v = bilin(feat + (size_t)ch * HW_, HF_, WF_, fx, fy);
  size_t mi = (size_t)r * 49 + (size_t)(py >> 1) * 7 + (px >> 1);
  Aconv[mi * 1024 + ch * 4 + (py & 1) * 2 + (px & 1)] = (h16)v;
}

// ------- generic WMMA GEMM: D = act(A[M][K] * B[Npad][K]^T + bias) ---------
// Block = 4 waves. Each wave computes a 32(M) x 64(N) macro-tile:
// 8 v_wmma per 32-wide K step, zero control flow inside the K loop.
// B must be row-padded so that every wave's 64 columns exist (zeros beyond N);
// only the epilogue stores are guarded by gn < N.
template <int MODE, bool RELU>
__global__ __launch_bounds__(128) void gemm_wmma(
    const h16* __restrict__ A, const h16* __restrict__ B,
    const float* __restrict__ bias, void* __restrict__ Dst,
    int M, int N, int K, int lda, int ldb, int ldd) {
  const int lane = threadIdx.x & 31;
  const int wave = threadIdx.x >> 5;
  const int hi = lane >> 4;      // 0 or 1 (lane half)
  const int lo = lane & 15;

  const int m0 = blockIdx.x * 32;
  const int nbase = (blockIdx.y * 4 + wave) * 64;

  // A fragment addressing: lane half hi holds K offsets {hi*8.., 16+hi*8..}
  const h16* __restrict__ ar0 = A + (size_t)(m0 + lo) * lda + hi * 8;
  const h16* __restrict__ ar1 = A + (size_t)(m0 + 16 + lo) * lda + hi * 8;
  // B fragment: lane holds one N column, K offsets {hi*8.., 16+hi*8..}
  const h16* __restrict__ bp0 = B + (size_t)(nbase + 0 + lo) * ldb + hi * 8;
  const h16* __restrict__ bp1 = B + (size_t)(nbase + 16 + lo) * ldb + hi * 8;
  const h16* __restrict__ bp2 = B + (size_t)(nbase + 32 + lo) * ldb + hi * 8;
  const h16* __restrict__ bp3 = B + (size_t)(nbase + 48 + lo) * ldb + hi * 8;

  v8f acc[2][4];
#pragma unroll
  for (int mi = 0; mi < 2; ++mi)
#pragma unroll
    for (int t = 0; t < 4; ++t) acc[mi][t] = (v8f){};

  for (int k = 0; k < K; k += 32) {
    // CDNA5 prefetch hint (global_prefetch_b8) one K-step ahead
    __builtin_prefetch(ar0 + k + 32, 0, 0);
    __builtin_prefetch(bp0 + k + 32, 0, 0);

    v16h a0 = load_afrag(ar0 + k);
    v16h a1 = load_afrag(ar1 + k);
    v16h b0 = load_afrag(bp0 + k);
    v16h b1 = load_afrag(bp1 + k);
    v16h b2 = load_afrag(bp2 + k);
    v16h b3 = load_afrag(bp3 + k);

    acc[0][0] = wmma_f16(a0, b0, acc[0][0]);
    acc[0][1] = wmma_f16(a0, b1, acc[0][1]);
    acc[0][2] = wmma_f16(a0, b2, acc[0][2]);
    acc[0][3] = wmma_f16(a0, b3, acc[0][3]);
    acc[1][0] = wmma_f16(a1, b0, acc[1][0]);
    acc[1][1] = wmma_f16(a1, b1, acc[1][1]);
    acc[1][2] = wmma_f16(a1, b2, acc[1][2]);
    acc[1][3] = wmma_f16(a1, b3, acc[1][3]);
  }

  // Epilogue. C/D layout: VGPR v, lanes 0-15 -> M=v, N=lane; lanes 16-31 -> M=v+8.
#pragma unroll
  for (int mi = 0; mi < 2; ++mi) {
#pragma unroll
    for (int t = 0; t < 4; ++t) {
      int gn = nbase + t * 16 + lo;
      if (gn < N) {
        float bv = bias[gn];
#pragma unroll
        for (int v = 0; v < 8; ++v) {
          int gm = m0 + mi * 16 + v + 8 * hi;
          float val = acc[mi][t][v] + bv;
          if (RELU) val = fmaxf(val, 0.0f);
          if constexpr (MODE == MODE_ROW_F16) {
            ((h16*)Dst)[(size_t)gm * ldd + gn] = (h16)val;
          } else if constexpr (MODE == MODE_CONV) {
            // gm = r*49 + p ; store x1[r][gn*49 + p] (FC1 flatten order c*49+p)
            int r = gm / 49, p = gm % 49;
            ((h16*)Dst)[(size_t)r * FCIN_ + (size_t)gn * 49 + p] = (h16)val;
          } else if constexpr (MODE == MODE_ROW_F32) {
            ((float*)Dst)[(size_t)gm * ldd + gn] = val;
          } else {  // MODE_OUT: gm = r*196 + j ; out[r][cls][j]
            int r = gm / NPTS_, j = gm % NPTS_;
            ((float*)Dst)[((size_t)r * NCLS_ + gn) * NPTS_ + j] = val;
          }
        }
      }
    }
  }
}

// ---------------- K6: uncertainty + exact top-k (147) point selection ------
__global__ __launch_bounds__(256) void select_points_kernel(
    const float* __restrict__ coarse, const float* __restrict__ cand,
    const float* __restrict__ cand2, const int* __restrict__ gtc,
    float* __restrict__ pts) {
  __shared__ float s_unc[NSAMP_];
  __shared__ float s_bv[256];
  __shared__ int s_bi[256];
  int r = blockIdx.x;
  int t = threadIdx.x;
  int gt = gtc[r];
  const float* pl = coarse + (size_t)r * PREDN_ + (size_t)gt * 49;
  for (int i = t; i < NSAMP_; i += 256) {
    float cx = cand[((size_t)r * NSAMP_ + i) * 2 + 0];
    float cy = cand[((size_t)r * NSAMP_ + i) * 2 + 1];
    float v = bilin(pl, OUTRES_, OUTRES_, cx * 7.0f - 0.5f, cy * 7.0f - 0.5f);
    s_unc[i] = -fabsf(v);
  }
  __syncthreads();
  // lax.top_k order: max first, ties -> lower index
  for (int it = 0; it < NUNC_; ++it) {
    float bv = -3.0e38f;
    int bi = 0x7fffffff;
    for (int i = t; i < NSAMP_; i += 256) {
      float v = s_unc[i];
      if (v > bv) { bv = v; bi = i; }     // ascending i => keep-first on tie
    }
    s_bv[t] = bv;
    s_bi[t] = bi;
    __syncthreads();
    if (t == 0) {
      float best = s_bv[0];
      int bidx = s_bi[0];
      for (int q = 1; q < 256; ++q) {
        if (s_bv[q] > best || (s_bv[q] == best && s_bi[q] < bidx)) {
          best = s_bv[q];
          bidx = s_bi[q];
        }
      }
      s_unc[bidx] = -3.4e38f;
      pts[((size_t)r * NPTS_ + it) * 2 + 0] = cand[((size_t)r * NSAMP_ + bidx) * 2 + 0];
      pts[((size_t)r * NPTS_ + it) * 2 + 1] = cand[((size_t)r * NSAMP_ + bidx) * 2 + 1];
    }
    __syncthreads();
  }
  if (t < NRAND_) {
    pts[((size_t)r * NPTS_ + NUNC_ + t) * 2 + 0] = cand2[((size_t)r * NRAND_ + t) * 2 + 0];
    pts[((size_t)r * NPTS_ + NUNC_ + t) * 2 + 1] = cand2[((size_t)r * NRAND_ + t) * 2 + 1];
  }
}

// ---------------- K7: fine features + coarse-at-points -> point-head input -
// phA[m][0..255] = fine ; phA/phB[m][256..335] = coarse_pts ; [336..351] = 0
__global__ __launch_bounds__(256) void build_ph_input_kernel(
    const float* __restrict__ feat, const float* __restrict__ boxes,
    const float* __restrict__ coarse, const float* __restrict__ pts,
    h16* __restrict__ phA, h16* __restrict__ phB) {
  int m = blockIdx.x;  // r*196 + j
  int r = m / NPTS_;
  int t = threadIdx.x;
  float ptx = pts[(size_t)m * 2 + 0];
  float pty = pts[(size_t)m * 2 + 1];
  float bx0 = boxes[r * 4 + 0], by0 = boxes[r * 4 + 1];
  float bx1 = boxes[r * 4 + 2], by1 = boxes[r * 4 + 3];
  float fx = (bx0 + ptx * (bx1 - bx0)) * 0.25f - 0.5f;
  float fy = (by0 + pty * (by1 - by0)) * 0.25f - 0.5f;
  float v = bilin(feat + (size_t)t * HW_, HF_, WF_, fx, fy);
  phA[(size_t)m * PHINP_ + t] = (h16)v;
  if (t < NCLS_) {
    float c = bilin(coarse + (size_t)r * PREDN_ + (size_t)t * 49, OUTRES_, OUTRES_,
                    ptx * 7.0f - 0.5f, pty * 7.0f - 0.5f);
    h16 ch = (h16)c;
    phA[(size_t)m * PHINP_ + 256 + t] = ch;
    phB[(size_t)m * PHINP_ + 256 + t] = ch;
  } else if (t < 96) {
    phA[(size_t)m * PHINP_ + PHIN_ + (t - 80)] = (h16)0.0f;
    phB[(size_t)m * PHINP_ + PHIN_ + (t - 80)] = (h16)0.0f;
  }
}

// ===========================================================================
extern "C" void kernel_launch(void* const* d_in, const int* in_sizes, int n_in,
                              void* d_out, int out_size, void* d_ws, size_t ws_size,
                              hipStream_t stream) {
  const float* features = (const float*)d_in[0];
  const float* boxes = (const float*)d_in[1];
  const float* rand_coords = (const float*)d_in[2];
  const float* rand_coords2 = (const float*)d_in[3];
  const float* conv_w = (const float*)d_in[4];
  const float* conv_b = (const float*)d_in[5];
  const float* fc1_w = (const float*)d_in[6];
  const float* fc1_b = (const float*)d_in[7];
  const float* fc2_w = (const float*)d_in[8];
  const float* fc2_b = (const float*)d_in[9];
  const float* pred_w = (const float*)d_in[10];
  const float* pred_b = (const float*)d_in[11];
  const float* ph_w1 = (const float*)d_in[12];
  const float* ph_b1 = (const float*)d_in[13];
  const float* ph_w2 = (const float*)d_in[14];
  const float* ph_b2 = (const float*)d_in[15];
  const float* ph_w3 = (const float*)d_in[16];
  const float* ph_b3 = (const float*)d_in[17];
  const float* ph_predw = (const float*)d_in[18];
  const float* ph_predb = (const float*)d_in[19];
  const int* gt_classes = (const int*)d_in[20];
  float* out = (float*)d_out;

  // ---- workspace bump allocator (256B aligned) ----
  char* ws = (char*)d_ws;
  size_t off = 0;
  auto alloc = [&](size_t bytes) -> void* {
    void* p = ws + off;
    off += (bytes + 255) & ~(size_t)255;
    return p;
  };
  h16* convW16 = (h16*)alloc((size_t)C_ * 1024 * 2);
  h16* fc1W16 = (h16*)alloc((size_t)FCDIM_ * FCIN_ * 2);
  h16* fc2W16 = (h16*)alloc((size_t)FCDIM_ * FCDIM_ * 2);
  h16* predW16 = (h16*)alloc((size_t)PREDNP_ * FCDIM_ * 2);   // row-padded
  h16* phW1_16 = (h16*)alloc((size_t)256 * PHINP_ * 2);
  h16* phW2_16 = (h16*)alloc((size_t)256 * PHINP_ * 2);
  h16* phW3_16 = (h16*)alloc((size_t)256 * PHINP_ * 2);
  h16* phPW_16 = (h16*)alloc((size_t)256 * PHINP_ * 2);       // 80 -> 256 rows
  h16* Aconv = (h16*)alloc((size_t)R_ * 49 * 1024 * 2);
  h16* x1 = (h16*)alloc((size_t)R_ * FCIN_ * 2);
  h16* h1b = (h16*)alloc((size_t)R_ * FCDIM_ * 2);
  h16* h2b = (h16*)alloc((size_t)R_ * FCDIM_ * 2);
  float* coarse = (float*)alloc((size_t)R_ * PREDN_ * 4);
  float* pts = (float*)alloc((size_t)R_ * NPTS_ * 2 * 4);
  h16* phA = (h16*)alloc((size_t)PHM_ * PHINP_ * 2);
  h16* phB = (h16*)alloc((size_t)PHM_ * PHINP_ * 2);

  auto cdiv = [](long a, long b) { return (unsigned)((a + b - 1) / b); };

  // ---- weight conversion to f16 (zero pad rows/cols as needed) ----
  {
    long n;
    n = (long)C_ * 1024;
    cvt_pad_kernel<<<cdiv(n, 256), 256, 0, stream>>>(conv_w, convW16, C_, 1024, 1024, n);
    n = (long)FCDIM_ * FCIN_;
    cvt_pad_kernel<<<cdiv(n, 256), 256, 0, stream>>>(fc1_w, fc1W16, FCDIM_, FCIN_, FCIN_, n);
    n = (long)FCDIM_ * FCDIM_;
    cvt_pad_kernel<<<cdiv(n, 256), 256, 0, stream>>>(fc2_w, fc2W16, FCDIM_, FCDIM_, FCDIM_, n);
    n = (long)PREDNP_ * FCDIM_;
    cvt_pad_kernel<<<cdiv(n, 256), 256, 0, stream>>>(pred_w, predW16, PREDN_, FCDIM_, FCDIM_, n);
    n = (long)256 * PHINP_;
    cvt_pad_kernel<<<cdiv(n, 256), 256, 0, stream>>>(ph_w1, phW1_16, 256, PHIN_, PHINP_, n);
    cvt_pad_kernel<<<cdiv(n, 256), 256, 0, stream>>>(ph_w2, phW2_16, 256, PHIN_, PHINP_, n);
    cvt_pad_kernel<<<cdiv(n, 256), 256, 0, stream>>>(ph_w3, phW3_16, 256, PHIN_, PHINP_, n);
    cvt_pad_kernel<<<cdiv(n, 256), 256, 0, stream>>>(ph_predw, phPW_16, NCLS_, PHIN_, PHINP_, n);
  }

  // ---- ROI pooling directly into im2col layout ----
  roi_im2col_kernel<<<R_ * POOL_ * POOL_, 256, 0, stream>>>(features, boxes, Aconv);

  // ---- conv as GEMM: M=25088, N=256, K=1024 (bias+ReLU, scatter to FC1 order)
  gemm_wmma<MODE_CONV, true><<<dim3(25088 / 32, 1), 128, 0, stream>>>(
      Aconv, convW16, conv_b, x1, 25088, 256, 1024, 1024, 1024, 0);

  // ---- FC1: M=512, N=1024, K=12544 ----
  gemm_wmma<MODE_ROW_F16, true><<<dim3(512 / 32, 4), 128, 0, stream>>>(
      x1, fc1W16, fc1_b, h1b, 512, FCDIM_, FCIN_, FCIN_, FCIN_, FCDIM_);

  // ---- FC2: M=512, N=1024, K=1024 ----
  gemm_wmma<MODE_ROW_F16, true><<<dim3(512 / 32, 4), 128, 0, stream>>>(
      h1b, fc2W16, fc2_b, h2b, 512, FCDIM_, FCDIM_, FCDIM_, FCDIM_, FCDIM_);

  // ---- pred: M=512, N=3920 (B padded to 4096), K=1024 -> coarse f32 ----
  gemm_wmma<MODE_ROW_F32, false><<<dim3(512 / 32, PREDNP_ / 256), 128, 0, stream>>>(
      h2b, predW16, pred_b, coarse, 512, PREDN_, FCDIM_, FCDIM_, FCDIM_, PREDN_);

  // ---- uncertainty-based exact top-k selection -> pts [512][196][2] ----
  select_points_kernel<<<R_, 256, 0, stream>>>(coarse, rand_coords, rand_coords2,
                                               gt_classes, pts);

  // ---- fine features + coarse@points -> point-head input (both buffers get
  //      the coarse columns + zero K-padding once; layers only rewrite 0..255)
  build_ph_input_kernel<<<PHM_, 256, 0, stream>>>(features, boxes, coarse, pts,
                                                  phA, phB);

  // ---- point head: 3x (M=100352, N=256, K=352) + pred (N=80, B padded 256)
  gemm_wmma<MODE_ROW_F16, true><<<dim3(PHM_ / 32, 1), 128, 0, stream>>>(
      phA, phW1_16, ph_b1, phB, PHM_, 256, PHINP_, PHINP_, PHINP_, PHINP_);
  gemm_wmma<MODE_ROW_F16, true><<<dim3(PHM_ / 32, 1), 128, 0, stream>>>(
      phB, phW2_16, ph_b2, phA, PHM_, 256, PHINP_, PHINP_, PHINP_, PHINP_);
  gemm_wmma<MODE_ROW_F16, true><<<dim3(PHM_ / 32, 1), 128, 0, stream>>>(
      phA, phW3_16, ph_b3, phB, PHM_, 256, PHINP_, PHINP_, PHINP_, PHINP_);
  gemm_wmma<MODE_OUT, false><<<dim3(PHM_ / 32, 1), 128, 0, stream>>>(
      phB, phPW_16, ph_predb, out, PHM_, NCLS_, PHINP_, PHINP_, PHINP_, 0);
}
